// AttentionLSTM_49100066128095
// MI455X (gfx1250) — matrix operand order repaired and checked by reading, not compiled, not run on hardware
//
#include <hip/hip_runtime.h>
#include <hip/hip_bf16.h>

// ---------------------------------------------------------------------------
// Shapes
// ---------------------------------------------------------------------------
#define B_   32
#define T_   128
#define P_   196
#define ENC_ 2048
#define DEC_ 512
#define ATT_ 512
#define EMB_ 512
#define V_   10000
#define XH_  3072   // EMB + ENC + DEC   (x = [emb | gate*awe], then h)
#define G4_  2048   // 4*DEC
#define PRE_ 2560   // ENC (f_beta rows) + ATT (dec_att rows)

typedef __attribute__((ext_vector_type(16))) __bf16        v16bf;
typedef __attribute__((ext_vector_type(8)))  float         v8f;
typedef __attribute__((ext_vector_type(4)))  unsigned int  v4u;

union BF16x16 {
    unsigned short u[16];
    v4u            q[2];
    v16bf          v;
};

__device__ __forceinline__ unsigned short f32_to_bf16(float f) {
    unsigned int x = __float_as_uint(f);
    unsigned int r = x + 0x7FFFu + ((x >> 16) & 1u);   // round-to-nearest-even
    return (unsigned short)(r >> 16);
}

__device__ __forceinline__ float sigmoidf(float x) { return 1.0f / (1.0f + __expf(-x)); }

// ---------------------------------------------------------------------------
// WMMA helpers (wave32, 16x16x32 bf16, f32 accumulate)
//
// A matrix 16x32 (MxK): lane L, elem i: M = L&15, K = (i&7) + (i>=8)*16 + (L>=16)*8
// B matrix 32x16 (KxN) = W^T of row-major W[N,K]: lane L, elem i: N = L&15, K = (i&7) + (i>=8)*16 + (L>=16)*8
// C/D 16x16 f32: elem j, lane L: M = j + (L>=16)*8, N = L&15
// ---------------------------------------------------------------------------
__device__ __forceinline__ v8f wmma_bf16(v16bf a, v16bf b, v8f c) {
    return __builtin_amdgcn_wmma_f32_16x16x32_bf16(
        false, a, false, b, (short)0, c, false, false);
}

// A tile from row-major bf16 activations
__device__ __forceinline__ v16bf load_A_bf16(const unsigned short* A, int lda, int m0, int k0, int lane) {
    int m = m0 + (lane & 15);
    const unsigned short* p = A + (size_t)m * lda + k0 + ((lane >> 4) << 3);
    BF16x16 r;
    r.q[0] = *(const v4u*)(p);        // K offsets +0..7
    r.q[1] = *(const v4u*)(p + 16);   // K offsets +16..23
    return r.v;
}

// B tile = W^T, W row-major bf16 (N_out x K)
__device__ __forceinline__ v16bf load_B_wT(const unsigned short* W, int ldw, int n0, int k0, int lane) {
    const unsigned short* p = W + (size_t)(n0 + (lane & 15)) * ldw + k0 + ((lane >> 4) << 3);
    BF16x16 r;
    r.q[0] = *(const v4u*)(p);
    r.q[1] = *(const v4u*)(p + 16);
    return r.v;
}

__device__ __forceinline__ void store_tile_bias(float* C, size_t ldc, int m0, int n0,
                                                int lane, v8f acc, float bias) {
    int n = n0 + (lane & 15);
    int mbase = m0 + ((lane >> 4) << 3);
#pragma unroll
    for (int j = 0; j < 8; ++j) C[(size_t)(mbase + j) * ldc + n] = acc[j] + bias;
}

// ---------------------------------------------------------------------------
// Kernel 1: conversions f32 -> bf16 (+ fused layouts)
//   enc_bf  : enc_att_W (512 x 2048)
//   wcat_bf : [W_ih | W_hh]   (2048 x 3072)
//   fc_bf   : fc_W (10000 x 512)
//   bias_g  : b_ih + b_hh (2048)
//   imgs_bf : bimgs (6272 x 2048)
//   wpre_bf : [f_beta_W ; dec_att_W]  (2560 x 512)
// ---------------------------------------------------------------------------
#define N_ENC_W  ((long long)ATT_ * ENC_)          // 1048576
#define N_WCAT   ((long long)G4_ * XH_)            // 6291456
#define N_FC     ((long long)V_ * DEC_)            // 5120000
#define N_IMG    ((long long)B_ * P_ * ENC_)       // 12845056
#define N_WPRE   ((long long)PRE_ * DEC_)          // 1310720
#define N_CONV   (N_ENC_W + N_WCAT + N_FC + G4_ + N_IMG + N_WPRE)

__global__ void k_convert(const float* __restrict__ enc_att_W,
                          const float* __restrict__ W_ih, const float* __restrict__ W_hh,
                          const float* __restrict__ fc_W,
                          const float* __restrict__ b_ih, const float* __restrict__ b_hh,
                          const float* __restrict__ imgs,
                          const float* __restrict__ f_beta_W, const float* __restrict__ dec_att_W,
                          unsigned short* __restrict__ enc_bf,
                          unsigned short* __restrict__ wcat_bf,
                          unsigned short* __restrict__ fc_bf,
                          float* __restrict__ bias_g,
                          unsigned short* __restrict__ imgs_bf,
                          unsigned short* __restrict__ wpre_bf) {
    long long idx = (long long)blockIdx.x * blockDim.x + threadIdx.x;
    if (idx < N_ENC_W) {
        enc_bf[idx] = f32_to_bf16(enc_att_W[idx]);
    } else if (idx < N_ENC_W + N_WCAT) {
        long long j = idx - N_ENC_W;
        int n = (int)(j / XH_), k = (int)(j % XH_);
        float v = (k < (EMB_ + ENC_)) ? W_ih[(size_t)n * (EMB_ + ENC_) + k]
                                      : W_hh[(size_t)n * DEC_ + (k - (EMB_ + ENC_))];
        wcat_bf[j] = f32_to_bf16(v);
    } else if (idx < N_ENC_W + N_WCAT + N_FC) {
        long long j = idx - (N_ENC_W + N_WCAT);
        fc_bf[j] = f32_to_bf16(fc_W[j]);
    } else if (idx < N_ENC_W + N_WCAT + N_FC + G4_) {
        int j = (int)(idx - (N_ENC_W + N_WCAT + N_FC));
        bias_g[j] = b_ih[j] + b_hh[j];
    } else if (idx < N_ENC_W + N_WCAT + N_FC + G4_ + N_IMG) {
        long long j = idx - (N_ENC_W + N_WCAT + N_FC + G4_);
        imgs_bf[j] = f32_to_bf16(imgs[j]);
    } else if (idx < N_CONV) {
        long long j = idx - (N_ENC_W + N_WCAT + N_FC + G4_ + N_IMG);
        int n = (int)(j / DEC_), k = (int)(j % DEC_);
        float v = (n < ENC_) ? f_beta_W[(size_t)n * DEC_ + k]
                             : dec_att_W[(size_t)(n - ENC_) * DEC_ + k];
        wpre_bf[j] = f32_to_bf16(v);
    }
}

// ---------------------------------------------------------------------------
// Kernel 2: h0/c0 init.  One block per batch element.
// ---------------------------------------------------------------------------
__global__ void k_init(const float* __restrict__ imgs,
                       const float* __restrict__ init_h_W, const float* __restrict__ init_h_b,
                       const float* __restrict__ init_c_W, const float* __restrict__ init_c_b,
                       float* __restrict__ h_ws, float* __restrict__ c_ws,
                       unsigned short* __restrict__ xh_bf) {
    __shared__ float mean_s[ENC_];
    int b = blockIdx.x, tid = threadIdx.x;
    const float* img_b = imgs + (size_t)b * P_ * ENC_;
    for (int e = tid; e < ENC_; e += blockDim.x) {
        float s = 0.f;
        for (int p = 0; p < P_; ++p) s += img_b[(size_t)p * ENC_ + e];
        mean_s[e] = s * (1.0f / P_);
    }
    __syncthreads();
    for (int j = tid; j < DEC_; j += blockDim.x) {
        float h = init_h_b[j], c = init_c_b[j];
        const float* wh = init_h_W + (size_t)j * ENC_;
        const float* wc = init_c_W + (size_t)j * ENC_;
        for (int k = 0; k < ENC_; ++k) { h += mean_s[k] * wh[k]; c += mean_s[k] * wc[k]; }
        h_ws[b * DEC_ + j] = h;
        c_ws[b * DEC_ + j] = c;
        xh_bf[(size_t)b * XH_ + (EMB_ + ENC_) + j] = f32_to_bf16(h);
    }
}

// ---------------------------------------------------------------------------
// Kernel 3: att1 = imgs @ enc_att_W.T + b   (6272 x 512, K=2048)  [WMMA 2x2]
// 3136 wave-jobs of 32x32 output each; 8 waves/block -> 392 blocks.
// ---------------------------------------------------------------------------
__global__ void k_att1(const unsigned short* __restrict__ imgs_bf,
                       const unsigned short* __restrict__ enc_bf,
                       const float* __restrict__ enc_att_b,
                       float* __restrict__ att1) {
    int lane = threadIdx.x & 31;
    int job = blockIdx.x * 8 + (threadIdx.x >> 5);    // 0..3135
    int mj = job >> 4, nj = job & 15;                 // 196 x 16
    int m0 = mj * 32, n0 = nj * 32;
    v8f a00 = {}, a01 = {}, a10 = {}, a11 = {};
    for (int k0 = 0; k0 < ENC_; k0 += 32) {
        v16bf fa0 = load_A_bf16(imgs_bf, ENC_, m0,      k0, lane);
        v16bf fa1 = load_A_bf16(imgs_bf, ENC_, m0 + 16, k0, lane);
        v16bf fb0 = load_B_wT(enc_bf, ENC_, n0,      k0, lane);
        v16bf fb1 = load_B_wT(enc_bf, ENC_, n0 + 16, k0, lane);
        a00 = wmma_bf16(fa0, fb0, a00);
        a10 = wmma_bf16(fa1, fb0, a10);
        a01 = wmma_bf16(fa0, fb1, a01);
        a11 = wmma_bf16(fa1, fb1, a11);
    }
    float bias0 = enc_att_b[n0 + (lane & 15)];
    float bias1 = enc_att_b[n0 + 16 + (lane & 15)];
    store_tile_bias(att1, ATT_, m0,      n0,      lane, a00, bias0);
    store_tile_bias(att1, ATT_, m0 + 16, n0,      lane, a10, bias0);
    store_tile_bias(att1, ATT_, m0,      n0 + 16, lane, a01, bias1);
    store_tile_bias(att1, ATT_, m0 + 16, n0 + 16, lane, a11, bias1);
}

// ---------------------------------------------------------------------------
// Kernel 4 (per step): pre = h @ [f_beta_W ; dec_att_W].T  (32 x 2560, K=512)
// [WMMA 2x4] 40 wave-jobs (2 Mt x 4 Nt each); 5 blocks.  Raw (bias added later).
// ---------------------------------------------------------------------------
__global__ void k_pre(const unsigned short* __restrict__ xh_bf,
                      const unsigned short* __restrict__ wpre_bf,
                      float* __restrict__ pre) {
    int lane = threadIdx.x & 31;
    int job = blockIdx.x * 8 + (threadIdx.x >> 5);
    if (job >= PRE_ / 64) return;                      // 40 jobs
    int n0 = job * 64;
    const unsigned short* Ah = xh_bf + (EMB_ + ENC_);  // h rows inside xh
    v8f acc[2][4] = {};
    for (int k0 = 0; k0 < DEC_; k0 += 32) {
        v16bf fa0 = load_A_bf16(Ah, XH_, 0,  k0, lane);
        v16bf fa1 = load_A_bf16(Ah, XH_, 16, k0, lane);
#pragma unroll
        for (int g = 0; g < 4; ++g) {
            v16bf fb = load_B_wT(wpre_bf, DEC_, n0 + 16 * g, k0, lane);
            acc[0][g] = wmma_bf16(fa0, fb, acc[0][g]);
            acc[1][g] = wmma_bf16(fa1, fb, acc[1][g]);
        }
    }
#pragma unroll
    for (int g = 0; g < 4; ++g) {
        store_tile_bias(pre, PRE_, 0,  n0 + 16 * g, lane, acc[0][g], 0.f);
        store_tile_bias(pre, PRE_, 16, n0 + 16 * g, lane, acc[1][g], 0.f);
    }
}

// ---------------------------------------------------------------------------
// Kernel 5 (per step): attention scores, softmax, awe, gate, build x.
// One block (256 thr) per batch element.  Uses pre[] for att2 & gate logits.
// ---------------------------------------------------------------------------
__global__ void k_step_attn(const float* __restrict__ imgs, const float* __restrict__ att1,
                            const float* __restrict__ pre,
                            const float* __restrict__ dec_att_b,
                            const float* __restrict__ full_att_W, const float* __restrict__ full_att_b,
                            const float* __restrict__ f_beta_b,
                            const float* __restrict__ emb_W,
                            const int* __restrict__ bx, const int* __restrict__ bxlen,
                            unsigned short* __restrict__ xh_bf,
                            float* __restrict__ alphas_out, int t) {
    __shared__ float att2_s[ATT_];
    __shared__ float e_s[P_];
    __shared__ float red_s[256];
    int b = blockIdx.x, tid = threadIdx.x;

    for (int a = tid; a < ATT_; a += 256)
        att2_s[a] = pre[(size_t)b * PRE_ + ENC_ + a] + dec_att_b[a];
    __syncthreads();

    // e[p] = relu(att1 + att2) . full_att_W + b
    for (int p = tid; p < P_; p += 256) {
        const float* ap = att1 + ((size_t)b * P_ + p) * ATT_;
        float s = full_att_b[0];
        for (int a = 0; a < ATT_; ++a) {
            float v = ap[a] + att2_s[a];
            s += (v > 0.f ? v : 0.f) * full_att_W[a];
        }
        e_s[p] = s;
    }
    __syncthreads();

    // softmax over P_
    float lm = -3.4e38f;
    for (int p = tid; p < P_; p += 256) lm = fmaxf(lm, e_s[p]);
    red_s[tid] = lm; __syncthreads();
    for (int s = 128; s > 0; s >>= 1) { if (tid < s) red_s[tid] = fmaxf(red_s[tid], red_s[tid + s]); __syncthreads(); }
    float mx = red_s[0]; __syncthreads();
    float ls = 0.f;
    for (int p = tid; p < P_; p += 256) ls += __expf(e_s[p] - mx);
    red_s[tid] = ls; __syncthreads();
    for (int s = 128; s > 0; s >>= 1) { if (tid < s) red_s[tid] += red_s[tid + s]; __syncthreads(); }
    float inv = 1.0f / red_s[0];
    __syncthreads();

    int active = (t < bxlen[b]) ? 1 : 0;
    for (int p = tid; p < P_; p += 256) {
        float al = __expf(e_s[p] - mx) * inv;
        e_s[p] = al;
        alphas_out[((size_t)b * T_ + t) * P_ + p] = active ? al : 0.0f;
    }
    __syncthreads();

    // awe (batched matvec over imgs) and gate (from pre logits); write x bf16
    const float* img_b = imgs + (size_t)b * P_ * ENC_;
#pragma unroll 1
    for (int r = 0; r < ENC_ / 256; ++r) {
        int e = tid + 256 * r;
        float awe = 0.f;
        for (int p = 0; p < P_; ++p) awe += e_s[p] * img_b[(size_t)p * ENC_ + e];
        float g = sigmoidf(pre[(size_t)b * PRE_ + e] + f_beta_b[e]);
        xh_bf[(size_t)b * XH_ + EMB_ + e] = f32_to_bf16(g * awe);
    }
    int tok = bx[b * T_ + t];
    for (int j = tid; j < EMB_; j += 256)
        xh_bf[(size_t)b * XH_ + j] = f32_to_bf16(emb_W[(size_t)tok * EMB_ + j]);
}

// ---------------------------------------------------------------------------
// Kernel 6 (per step): gates = [x|h] @ [W_ih|W_hh].T + bias  [WMMA 2x4]
// M=32 x N=2048, K=3072.  32 wave-jobs (2 Mt x 4 Nt); 4 blocks.
// ---------------------------------------------------------------------------
__global__ void k_gates(const unsigned short* __restrict__ xh_bf,
                        const unsigned short* __restrict__ wcat_bf,
                        const float* __restrict__ bias_g,
                        float* __restrict__ gates) {
    int lane = threadIdx.x & 31;
    int job = blockIdx.x * 8 + (threadIdx.x >> 5);     // 0..31
    int n0 = job * 64;
    v8f acc[2][4] = {};
    for (int k0 = 0; k0 < XH_; k0 += 32) {
        v16bf fa0 = load_A_bf16(xh_bf, XH_, 0,  k0, lane);
        v16bf fa1 = load_A_bf16(xh_bf, XH_, 16, k0, lane);
#pragma unroll
        for (int g = 0; g < 4; ++g) {
            v16bf fb = load_B_wT(wcat_bf, XH_, n0 + 16 * g, k0, lane);
            acc[0][g] = wmma_bf16(fa0, fb, acc[0][g]);
            acc[1][g] = wmma_bf16(fa1, fb, acc[1][g]);
        }
    }
#pragma unroll
    for (int g = 0; g < 4; ++g) {
        float bias = bias_g[n0 + 16 * g + (lane & 15)];
        store_tile_bias(gates, G4_, 0,  n0 + 16 * g, lane, acc[0][g], bias);
        store_tile_bias(gates, G4_, 16, n0 + 16 * g, lane, acc[1][g], bias);
    }
}

// ---------------------------------------------------------------------------
// Kernel 7 (per step): LSTM pointwise update with length masking.
// ---------------------------------------------------------------------------
__global__ void k_lstm(const float* __restrict__ gates,
                       float* __restrict__ h_ws, float* __restrict__ c_ws,
                       unsigned short* __restrict__ xh_bf,
                       const int* __restrict__ bxlen, int t) {
    int idx = blockIdx.x * blockDim.x + threadIdx.x;   // 0 .. B_*DEC_-1
    if (idx >= B_ * DEC_) return;
    int b = idx / DEC_, j = idx % DEC_;
    const float* g = gates + (size_t)b * G4_;
    float i_ = sigmoidf(g[j]);
    float f_ = sigmoidf(g[DEC_ + j]);
    float gg = tanhf(g[2 * DEC_ + j]);
    float o_ = sigmoidf(g[3 * DEC_ + j]);
    float c_old = c_ws[idx];
    float c_new = f_ * c_old + i_ * gg;
    float h_new = o_ * tanhf(c_new);
    bool act = t < bxlen[b];
    float h2 = act ? h_new : h_ws[idx];
    float c2 = act ? c_new : c_old;
    h_ws[idx] = h2;
    c_ws[idx] = c2;
    xh_bf[(size_t)b * XH_ + (EMB_ + ENC_) + j] = f32_to_bf16(h2);
}

// ---------------------------------------------------------------------------
// Kernel 8 (per step): preds = h @ fc_W.T + fc_b, masked -> d_out  [WMMA 2x5]
// M=32 x N=10000, K=512.  125 wave-jobs (2 Mt x 5 Nt); 16 blocks (guarded).
// ---------------------------------------------------------------------------
__global__ void k_preds(const unsigned short* __restrict__ xh_bf,
                        const unsigned short* __restrict__ fc_bf,
                        const float* __restrict__ fc_b,
                        const int* __restrict__ bxlen,
                        float* __restrict__ preds_out, int t) {
    int lane = threadIdx.x & 31;
    int job = blockIdx.x * 8 + (threadIdx.x >> 5);
    if (job >= V_ / 80) return;                        // 125 jobs
    int n0 = job * 80;
    const unsigned short* Ah = xh_bf + (EMB_ + ENC_);  // h rows inside xh
    v8f acc[2][5] = {};
    for (int k0 = 0; k0 < DEC_; k0 += 32) {
        v16bf fa0 = load_A_bf16(Ah, XH_, 0,  k0, lane);
        v16bf fa1 = load_A_bf16(Ah, XH_, 16, k0, lane);
#pragma unroll
        for (int g = 0; g < 5; ++g) {
            v16bf fb = load_B_wT(fc_bf, DEC_, n0 + 16 * g, k0, lane);
            acc[0][g] = wmma_bf16(fa0, fb, acc[0][g]);
            acc[1][g] = wmma_bf16(fa1, fb, acc[1][g]);
        }
    }
#pragma unroll
    for (int g = 0; g < 5; ++g) {
        int n = n0 + 16 * g + (lane & 15);
        float bias = fc_b[n];
#pragma unroll
        for (int mt = 0; mt < 2; ++mt) {
            int mbase = mt * 16 + ((lane >> 4) << 3);
#pragma unroll
            for (int j = 0; j < 8; ++j) {
                int b = mbase + j;                     // batch index (M == B_)
                bool act = t < bxlen[b];
                preds_out[((size_t)b * T_ + t) * V_ + n] =
                    act ? (acc[mt][g][j] + bias) : 0.0f;
            }
        }
    }
}

// ---------------------------------------------------------------------------
// Host launcher
// ---------------------------------------------------------------------------
extern "C" void kernel_launch(void* const* d_in, const int* in_sizes, int n_in,
                              void* d_out, int out_size, void* d_ws, size_t ws_size,
                              hipStream_t stream) {
    (void)in_sizes; (void)n_in; (void)out_size; (void)ws_size;
    const float* bimgs     = (const float*)d_in[0];
    const int*   bx        = (const int*)  d_in[1];
    const int*   bxlen     = (const int*)  d_in[2];
    const float* emb_W     = (const float*)d_in[3];
    const float* enc_att_W = (const float*)d_in[4];
    const float* enc_att_b = (const float*)d_in[5];
    const float* dec_att_W = (const float*)d_in[6];
    const float* dec_att_b = (const float*)d_in[7];
    const float* full_att_W= (const float*)d_in[8];
    const float* full_att_b= (const float*)d_in[9];
    const float* init_h_W  = (const float*)d_in[10];
    const float* init_h_b  = (const float*)d_in[11];
    const float* init_c_W  = (const float*)d_in[12];
    const float* init_c_b  = (const float*)d_in[13];
    const float* f_beta_W  = (const float*)d_in[14];
    const float* f_beta_b  = (const float*)d_in[15];
    const float* W_ih      = (const float*)d_in[16];
    const float* b_ih      = (const float*)d_in[17];
    const float* W_hh      = (const float*)d_in[18];
    const float* b_hh      = (const float*)d_in[19];
    const float* fc_W      = (const float*)d_in[20];
    const float* fc_b      = (const float*)d_in[21];

    float* preds_out  = (float*)d_out;                              // (B,T,V)
    float* alphas_out = (float*)d_out + (size_t)B_ * T_ * V_;       // (B,T,P)

    // workspace layout (256B aligned chunks)
    char* ws = (char*)d_ws;
    size_t off = 0;
    auto take = [&](size_t bytes) { char* p = ws + off; off += (bytes + 255) & ~(size_t)255; return p; };
    unsigned short* enc_bf  = (unsigned short*)take((size_t)N_ENC_W * 2);
    unsigned short* wcat_bf = (unsigned short*)take((size_t)N_WCAT * 2);
    unsigned short* fc_bf   = (unsigned short*)take((size_t)N_FC * 2);
    float*          bias_g  = (float*)take(G4_ * 4);
    unsigned short* imgs_bf = (unsigned short*)take((size_t)N_IMG * 2);
    unsigned short* wpre_bf = (unsigned short*)take((size_t)N_WPRE * 2);
    float*          att1    = (float*)take((size_t)B_ * P_ * ATT_ * 4);
    float*          pre     = (float*)take((size_t)B_ * PRE_ * 4);
    float*          h_ws    = (float*)take(B_ * DEC_ * 4);
    float*          c_ws    = (float*)take(B_ * DEC_ * 4);
    unsigned short* xh_bf   = (unsigned short*)take((size_t)B_ * XH_ * 2);
    float*          gates   = (float*)take((size_t)B_ * G4_ * 4);

    // --- precompute ---
    {
        long long blocks = (N_CONV + 255) / 256;
        k_convert<<<(int)blocks, 256, 0, stream>>>(enc_att_W, W_ih, W_hh, fc_W, b_ih, b_hh,
                                                   bimgs, f_beta_W, dec_att_W,
                                                   enc_bf, wcat_bf, fc_bf, bias_g,
                                                   imgs_bf, wpre_bf);
    }
    k_init<<<B_, 256, 0, stream>>>(bimgs, init_h_W, init_h_b, init_c_W, init_c_b,
                                   h_ws, c_ws, xh_bf);
    {
        int jobs = (B_ * P_ / 32) * (ATT_ / 32);   // 196 * 16 = 3136
        k_att1<<<jobs / 8, 256, 0, stream>>>(imgs_bf, enc_bf, enc_att_b, att1);
    }

    // --- sequential decode ---
    for (int t = 0; t < T_; ++t) {
        k_pre<<<5, 256, 0, stream>>>(xh_bf, wpre_bf, pre);
        k_step_attn<<<B_, 256, 0, stream>>>(bimgs, att1, pre,
                                            dec_att_b, full_att_W, full_att_b,
                                            f_beta_b, emb_W, bx, bxlen,
                                            xh_bf, alphas_out, t);
        k_gates<<<4, 256, 0, stream>>>(xh_bf, wcat_bf, bias_g, gates);
        k_lstm<<<(B_ * DEC_ + 255) / 256, 256, 0, stream>>>(gates, h_ws, c_ws, xh_bf, bxlen, t);
        k_preds<<<16, 256, 0, stream>>>(xh_bf, fc_bf, fc_b, bxlen, preds_out, t);
    }
}
